// PatchToEmbeddings_89541478187367
// MI455X (gfx1250) — matrix operand order repaired and checked by reading, not compiled, not run on hardware
//
#include <hip/hip_runtime.h>
#include <hip/hip_bf16.h>
#include <math.h>

// ---------------------------------------------------------------------------
// PatchToEmbeddings for MI455X (gfx1250, wave32, WMMA bf16, async LDS DMA)
//   out = (patch_classes [N] (written as float), similarity [N,512] fp32)
//   N = 825216, D = 96, E = 512, ED = 768
// ---------------------------------------------------------------------------

typedef __bf16 v16bf __attribute__((ext_vector_type(16)));
typedef float  v8f   __attribute__((ext_vector_type(8)));
typedef float  f4    __attribute__((ext_vector_type(4)));
typedef unsigned int u4 __attribute__((ext_vector_type(4)));

union BFrag { u4 u[2]; v16bf v; };

#define D_DIM 96
#define E_DIM 512
#define ED_DIM 768

// ---------------------------------------------------------------------------
// Kernel 1: pe[e][d] = l2norm_row(embeddings[e] @ W_embed + b_embed) -> bf16
// grid: 512 blocks x 96 threads
// ---------------------------------------------------------------------------
__global__ void pe_project_kernel(const float* __restrict__ emb,
                                  const float* __restrict__ W_embed,
                                  const float* __restrict__ b_embed,
                                  __bf16* __restrict__ pe) {
  const int e = blockIdx.x;
  const int d = threadIdx.x;
  __shared__ float red[D_DIM];
  __shared__ float ssum;

  float acc = b_embed[d];
  const float* er = emb + (long)e * ED_DIM;
  for (int k = 0; k < ED_DIM; ++k)
    acc = fmaf(er[k], W_embed[k * D_DIM + d], acc);

  red[d] = acc * acc;
  __syncthreads();
  if (d == 0) {
    float s = 0.f;
    for (int i = 0; i < D_DIM; ++i) s += red[i];
    ssum = s;
  }
  __syncthreads();
  const float scale = 1.0f / fmaxf(sqrtf(ssum), 1e-12f);
  pe[e * D_DIM + d] = (__bf16)(acc * scale);
}

// ---------------------------------------------------------------------------
// Kernel 2: Wt[n][k] = bf16(W_patch[k][n])   (B-fragment friendly layout)
// grid: 96 blocks x 96 threads
// ---------------------------------------------------------------------------
__global__ void wt_kernel(const float* __restrict__ W_patch,
                          __bf16* __restrict__ wt) {
  const int n = blockIdx.x;
  const int k = threadIdx.x;
  wt[n * D_DIM + k] = (__bf16)W_patch[k * D_DIM + n];
}

// ---------------------------------------------------------------------------
// Kernel 3: main fused kernel.
//  block = 256 threads = 8 waves; wave w handles rows [base .. base+16).
//  Codebook (96 KB bf16) is DMA'd into LDS with global_load_async_to_lds_b128
//  (ASYNCcnt) overlapped with the projection WMMAs.
// ---------------------------------------------------------------------------
__global__ __launch_bounds__(256)
void sim_kernel(const float* __restrict__ patches,
                const float* __restrict__ b_patch,
                const __bf16* __restrict__ wt,
                const __bf16* __restrict__ pe_g,
                float* __restrict__ out_class,
                float* __restrict__ out_sim) {
  __shared__ unsigned short s_pe[E_DIM * D_DIM];   // 96 KB bf16 codebook
  __shared__ float s_pp[8 * 16 * D_DIM];           // 48 KB pp staging (reused for argmax)
  __shared__ float s_scale[8 * 16];

  const int tid  = threadIdx.x;
  const int wave = tid >> 5;
  const int lane = tid & 31;
  const int m    = lane & 15;   // row (A) / column (B,C) within tile
  const int half = lane >> 4;   // K-half selector

  // ---- async DMA: whole bf16 codebook -> LDS (96 KB), overlapped with
  //      projection below; tracked by ASYNCcnt, fenced before first barrier.
  {
    // Low 32 bits of a flat shared-memory address are the LDS byte offset
    // (see ISA 10.2 aperture calculation).
    const unsigned lds_base = (unsigned)(unsigned long long)(const void*)s_pe;
    #pragma unroll
    for (int i = 0; i < (E_DIM * D_DIM * 2 / 16) / 256; ++i) {
      const int idx = tid + i * 256;                       // 16B chunk index
      const unsigned long long g =
          (unsigned long long)(const void*)((const char*)pe_g + idx * 16);
      const unsigned l = lds_base + idx * 16;
      asm volatile("global_load_async_to_lds_b128 %0, %1, off"
                   :: "v"(l), "v"(g)
                   : "memory");
    }
  }

  const long tile_base = ((long)blockIdx.x * 8 + wave) * 16;

  // ---- projection: C[6] = patches_tile(16x96) @ W_patch(96x96) ----
  v8f c[6];
  #pragma unroll
  for (int t = 0; t < 6; ++t) c[t] = (v8f){};

  const float* __restrict__ arow = patches + (tile_base + m) * D_DIM;
  #pragma unroll
  for (int ks = 0; ks < 3; ++ks) {
    // A fragment (16x32 bf16): lane m, contiguous K [k0..k0+15]
    const int k0 = ks * 32 + half * 16;
    f4 g0 = *(const f4*)(arow + k0);
    f4 g1 = *(const f4*)(arow + k0 + 4);
    f4 g2 = *(const f4*)(arow + k0 + 8);
    f4 g3 = *(const f4*)(arow + k0 + 12);
    v16bf a;
    #pragma unroll
    for (int e2 = 0; e2 < 4; ++e2) {
      a[e2]      = (__bf16)g0[e2];
      a[4 + e2]  = (__bf16)g1[e2];
      a[8 + e2]  = (__bf16)g2[e2];
      a[12 + e2] = (__bf16)g3[e2];
    }
    #pragma unroll
    for (int t = 0; t < 6; ++t) {
      // B fragment of W: lane col n = t*16+m, contiguous 16 bf16 K values
      BFrag bw;
      const u4* wp = (const u4*)(wt + (long)(t * 16 + m) * D_DIM + ks * 32 + half * 16);
      bw.u[0] = wp[0];
      bw.u[1] = wp[1];
      c[t] = __builtin_amdgcn_wmma_f32_16x16x32_bf16(
                 false, a, false, bw.v, (short)0, c[t], false, false);
    }
  }

  // ---- bias add, stage pp rows to LDS ----
  float* pp = s_pp + wave * 16 * D_DIM;
  #pragma unroll
  for (int t = 0; t < 6; ++t) {
    const float bias = b_patch[t * 16 + m];
    #pragma unroll
    for (int r = 0; r < 8; ++r)
      pp[(r + half * 8) * D_DIM + t * 16 + m] = c[t][r] + bias;
  }

  // codebook DMA must be complete before any wave passes this barrier
  asm volatile("s_wait_asynccnt 0x0" ::: "memory");
  __syncthreads();

  // ---- per-row L2 norm -> scale ----
  if (lane < 16) {
    const float* row = pp + lane * D_DIM;
    float ss = 0.f;
    #pragma unroll
    for (int k = 0; k < D_DIM; ++k) ss = fmaf(row[k], row[k], ss);
    s_scale[wave * 16 + lane] = 1.0f / fmaxf(sqrtf(ss), 1e-12f);
  }
  __syncthreads();

  // ---- build normalized bf16 A fragments for similarity ----
  v16bf as[3];
  {
    const float sc = s_scale[wave * 16 + m];
    const float* row = pp + m * D_DIM;
    #pragma unroll
    for (int ks = 0; ks < 3; ++ks) {
      const int k0 = ks * 32 + half * 16;
      f4 g0 = *(const f4*)(row + k0);
      f4 g1 = *(const f4*)(row + k0 + 4);
      f4 g2 = *(const f4*)(row + k0 + 8);
      f4 g3 = *(const f4*)(row + k0 + 12);
      v16bf a;
      #pragma unroll
      for (int e2 = 0; e2 < 4; ++e2) {
        a[e2]      = (__bf16)(g0[e2] * sc);
        a[4 + e2]  = (__bf16)(g1[e2] * sc);
        a[8 + e2]  = (__bf16)(g2[e2] * sc);
        a[12 + e2] = (__bf16)(g3[e2] * sc);
      }
      as[ks] = a;
    }
  }

  // ---- similarity vs all 512 codes: 32 E-tiles x 3 K-steps ----
  float bestv[8];
  int   besti[8];
  #pragma unroll
  for (int r = 0; r < 8; ++r) { bestv[r] = -3.0e38f; besti[r] = 0; }

  for (int et = 0; et < 32; ++et) {
    v8f cs = (v8f){};
    #pragma unroll
    for (int ks = 0; ks < 3; ++ks) {
      BFrag bp;
      const u4* sp = (const u4*)(s_pe + ((et * 16 + m) * D_DIM + ks * 32 + half * 16));
      bp.u[0] = sp[0];
      bp.u[1] = sp[1];
      cs = __builtin_amdgcn_wmma_f32_16x16x32_bf16(
               false, as[ks], false, bp.v, (short)0, cs, false, false);
    }
    const int col = et * 16 + m;
    #pragma unroll
    for (int r = 0; r < 8; ++r) {
      const float v = cs[r];
      out_sim[(tile_base + r + half * 8) * E_DIM + col] = v;
      if (v > bestv[r]) { bestv[r] = v; besti[r] = col; }
    }
  }

  // ---- cross-lane argmax (reuse s_pp region as staging) ----
  __syncthreads();
  float* sv = s_pp;                    // 8 waves * 8 * 32 floats
  int*   si = (int*)(s_pp + 2048);     // 8 waves * 8 * 32 ints
  float* wv = sv + wave * 256;
  int*   wi = si + wave * 256;
  #pragma unroll
  for (int r = 0; r < 8; ++r) {
    wv[r * 32 + lane] = bestv[r];
    wi[r * 32 + lane] = besti[r];
  }
  __syncthreads();
  if (lane < 16) {
    const int row = lane;
    const int r  = row & 7;
    const int l0 = (row < 8) ? 0 : 16;
    float bv = -3.0e38f;
    int   bi = 0x7fffffff;
    for (int l = l0; l < l0 + 16; ++l) {
      const float v = wv[r * 32 + l];
      const int   i = wi[r * 32 + l];
      if (v > bv || (v == bv && i < bi)) { bv = v; bi = i; }
    }
    out_class[tile_base + row] = (float)bi;
  }
}

// ---------------------------------------------------------------------------
extern "C" void kernel_launch(void* const* d_in, const int* in_sizes, int n_in,
                              void* d_out, int out_size, void* d_ws, size_t ws_size,
                              hipStream_t stream) {
  const float* patches    = (const float*)d_in[0];
  const float* embeddings = (const float*)d_in[1];
  const float* W_patch    = (const float*)d_in[2];
  const float* b_patch    = (const float*)d_in[3];
  const float* W_embed    = (const float*)d_in[4];
  const float* b_embed    = (const float*)d_in[5];

  __bf16* pe = (__bf16*)d_ws;                                  // 512*96 bf16
  __bf16* wt = (__bf16*)((char*)d_ws + E_DIM * D_DIM * 2);     // 96*96 bf16

  const long n_rows = (long)in_sizes[0] / D_DIM;               // 825216
  float* out_class = (float*)d_out;                            // [N] classes (as float)
  float* out_sim   = out_class + n_rows;                       // [N,512] similarity

  pe_project_kernel<<<E_DIM, D_DIM, 0, stream>>>(embeddings, W_embed, b_embed, pe);
  wt_kernel<<<D_DIM, D_DIM, 0, stream>>>(W_patch, wt);

  const int blocks = (int)(n_rows / 128);                      // 6447, exact
  sim_kernel<<<blocks, 256, 0, stream>>>(patches, b_patch, wt, pe, out_class, out_sim);
}
